// APG_MLP_Layer_32530082299937
// MI455X (gfx1250) — matrix operand order repaired and checked by reading, not compiled, not run on hardware
//
#include <hip/hip_runtime.h>

// ---------------------------------------------------------------------------
// APG hyper-net MLP, fused per-layer kernel for gfx1250 (MI455X).
//   h[b,s] = sum_r u[b,r] * ( sum_k x[b,k]*Wm[r*64+s,k] + bm[r*64+s] )
//   out    = relu( h @ V^T + b )
// Never materializes the [8192,4096] per-sample weight tensor (saves ~800MB
// of HBM traffic per launch). bf16 WMMA (16x16x32) with f32 accumulate.
// ---------------------------------------------------------------------------

typedef unsigned short u16;
typedef unsigned int   u32;

typedef __attribute__((ext_vector_type(16))) __bf16 v16bf;
typedef __attribute__((ext_vector_type(8)))  float  v8f;

struct alignas(16) B128 { u32 x, y, z, w; };
union  FragU { B128 q[2]; v16bf v; };

#define BATCH  8192
#define DIN    1024
#define DOUT   1024
#define RANK   64
#define TILE_M 64     // samples per workgroup (4 waves x 16 rows)

static __device__ __forceinline__ v16bf frag_ld(const u16* p0, const u16* p1) {
  FragU f;
  f.q[0] = *(const B128*)p0;   // 16B -> global_load_b128 / ds_load_b128
  f.q[1] = *(const B128*)p1;
  return f.v;
}

static __device__ __forceinline__ v8f wmma_bf16(v16bf a, v16bf b, v8f c) {
  return __builtin_amdgcn_wmma_f32_16x16x32_bf16(false, a, false, b,
                                                 (short)0, c, false, false);
}

static __device__ __forceinline__ u16 f2bf(float f) {   // round-to-nearest-even
  u32 u = __float_as_uint(f);
  u32 r = (u + 0x7FFFu + ((u >> 16) & 1u)) >> 16;
  return (u16)r;
}

// ---------------- fp32 -> bf16 pack (8 elems / thread) ----------------------
__global__ __launch_bounds__(256)
void k_cvt_bf16(const float* __restrict__ in, u16* __restrict__ out, int n8) {
  int i = blockIdx.x * 256 + threadIdx.x;
  if (i >= n8) return;
  const float4* p = (const float4*)(in + (size_t)i * 8);
  float4 f0 = p[0], f1 = p[1];
  B128 s;
  s.x = (u32)f2bf(f0.x) | ((u32)f2bf(f0.y) << 16);
  s.y = (u32)f2bf(f0.z) | ((u32)f2bf(f0.w) << 16);
  s.z = (u32)f2bf(f1.x) | ((u32)f2bf(f1.y) << 16);
  s.w = (u32)f2bf(f1.z) | ((u32)f2bf(f1.w) << 16);
  *(B128*)(out + (size_t)i * 8) = s;
}

// ---------------- fused layer --------------------------------------------
// grid: BATCH/TILE_M = 128 workgroups, 128 threads (4 waves).
// Each wave owns 16 sample rows; A fragments come straight from row-major
// activations, B fragments straight from row-major [out,in] weights (a B
// "column" == a weight row, contiguous K per lane -> no transposes).
__global__ __launch_bounds__(128, 1)
void k_apg_layer(const u16*  __restrict__ xb,     // [B, DIN] bf16 activation
                 const u16*  __restrict__ Wmb,    // [4096, DIN] bf16
                 const float* __restrict__ bm,    // [4096]
                 const u16*  __restrict__ Ub,     // [64, DIN] bf16
                 const u16*  __restrict__ Vb,     // [DOUT, 64] bf16
                 const float* __restrict__ bias,  // [DOUT]
                 u16*  __restrict__ out_bf,       // next activation (bf16) or null
                 float* __restrict__ out_f32,     // final output (f32) or null
                 int activate)
{
  __shared__ float bm_lds[4096];       // 16 KB
  __shared__ float u_lds[64 * 64];     // 16 KB  u[m_local][r]
  __shared__ u16   h_lds[64 * 72];     // 9 KB   h tile (pad 8 -> 16B-aligned rows)

  const int tid  = threadIdx.x;
  const int wave = tid >> 5;
  const int lane = tid & 31;
  const int l15  = lane & 15;
  const int hi   = lane >> 4;          // which half-wave
  const int mh   = hi << 3;            // C/D row offset: m = v + mh
  const int m0   = blockIdx.x * TILE_M;

  for (int i = tid; i < 4096; i += 128) bm_lds[i] = bm[i];
  __syncthreads();

  // Per-lane fragment base pointers.
  // A (16-bit, 16x32): lane reads row m=l15, K chunks {hi*8 .. +7} and {+16..+23}
  // B (16-bit, 32x16): lane reads weight row (=B column), K chunks {hi*8 .. +7} and {+16..+23}
  const int   myrow = m0 + wave * 16 + l15;
  const u16* px = xb  + (size_t)myrow * DIN + (hi ? 8 : 0);
  const u16* pw = Wmb + (size_t)l15   * DIN + (hi ? 8 : 0);
  const u16* pu = Ub  + (size_t)l15   * DIN + (hi ? 8 : 0);

  v8f vzero = {0.f,0.f,0.f,0.f,0.f,0.f,0.f,0.f};

  // ---- u = x @ U^T  (16 x 64 per wave) ----
  v8f uacc[4];
  #pragma unroll
  for (int nt = 0; nt < 4; ++nt) uacc[nt] = vzero;
  for (int k = 0; k < DIN; k += 32) {
    v16bf a = frag_ld(px + k, px + k + 16);
    #pragma unroll
    for (int nt = 0; nt < 4; ++nt) {
      const u16* pb = pu + (size_t)(nt * 16) * DIN + k;
      uacc[nt] = wmma_bf16(a, frag_ld(pb, pb + 16), uacc[nt]);
    }
  }
  #pragma unroll
  for (int nt = 0; nt < 4; ++nt)
    #pragma unroll
    for (int v = 0; v < 8; ++v)
      u_lds[(wave * 16 + v + mh) * 64 + (nt * 16 + l15)] = uacc[nt][v];
  // (each wave reads back only its own rows -> no barrier needed)

  // ---- h[m][s] = sum_r u[m][r] * (x @ Wm_block(r)^T + bm) ----
  v8f h[4];
  #pragma unroll
  for (int nt = 0; nt < 4; ++nt) h[nt] = vzero;

  for (int rg = 0; rg < RANK; rg += 2) {            // 2 r-values share each A frag
    v8f acc0[4], acc1[4];
    #pragma unroll
    for (int nt = 0; nt < 4; ++nt) { acc0[nt] = vzero; acc1[nt] = vzero; }

    if (rg + 2 < RANK)                               // warm L2/WGP$ for next block
      __builtin_prefetch(pw + (size_t)((rg + 2) * 64) * DIN, 0, 1);

    for (int k = 0; k < DIN; k += 32) {
      v16bf a = frag_ld(px + k, px + k + 16);
      const u16* pb0 = pw + (size_t)(rg * 64) * DIN + k;
      #pragma unroll
      for (int nt = 0; nt < 4; ++nt) {
        const u16* pb = pb0 + (size_t)(nt * 16) * DIN;
        acc0[nt] = wmma_bf16(a, frag_ld(pb, pb + 16), acc0[nt]);
      }
      const u16* pb1 = pb0 + (size_t)64 * DIN;
      #pragma unroll
      for (int nt = 0; nt < 4; ++nt) {
        const u16* pb = pb1 + (size_t)(nt * 16) * DIN;
        acc1[nt] = wmma_bf16(a, frag_ld(pb, pb + 16), acc1[nt]);
      }
    }
    #pragma unroll
    for (int j = 0; j < 2; ++j) {
      const int r = rg + j;
      float uv[8];
      #pragma unroll
      for (int v = 0; v < 8; ++v)
        uv[v] = u_lds[(wave * 16 + v + mh) * 64 + r];   // 2-way broadcast read
      v8f* acc = j ? acc1 : acc0;
      #pragma unroll
      for (int nt = 0; nt < 4; ++nt) {
        float bmv = bm_lds[r * 64 + nt * 16 + l15];
        #pragma unroll
        for (int v = 0; v < 8; ++v)
          h[nt][v] += uv[v] * (acc[nt][v] + bmv);
      }
    }
  }

  // ---- stage h as bf16 in LDS, then out = relu(h @ V^T + b) ----
  #pragma unroll
  for (int nt = 0; nt < 4; ++nt)
    #pragma unroll
    for (int v = 0; v < 8; ++v)
      h_lds[(wave * 16 + v + mh) * 72 + nt * 16 + l15] = f2bf(h[nt][v]);
  __syncthreads();

  const int  jbase = wave * 256;                       // waves split DOUT
  const u16* pv = Vb + (size_t)(jbase + l15) * RANK + (hi ? 8 : 0);
  const u16* ph = &h_lds[l15 * 72 + (hi ? 8 : 0)];
  for (int mt = 0; mt < 4; ++mt) {
    for (int jt = 0; jt < 16; ++jt) {
      v8f acc = vzero;
      #pragma unroll
      for (int kk = 0; kk < RANK; kk += 32) {
        const u16* pa = ph + (size_t)(mt * 16) * 72 + kk;   // ds_load_b128 x2
        const u16* pb = pv + (size_t)(jt * 16) * RANK + kk;
        acc = wmma_bf16(frag_ld(pa, pa + 16), frag_ld(pb, pb + 16), acc);
      }
      const int jcol = jbase + jt * 16 + l15;
      const float bj = bias[jcol];
      #pragma unroll
      for (int v = 0; v < 8; ++v) {
        float val = acc[v] + bj;
        if (activate) val = fmaxf(val, 0.0f);
        const size_t oi = (size_t)(m0 + mt * 16 + v + mh) * DOUT + jcol;
        if (out_f32) out_f32[oi] = val;
        else         out_bf[oi]  = f2bf(val);
      }
    }
  }
}

// ---------------------------------------------------------------------------
extern "C" void kernel_launch(void* const* d_in, const int* in_sizes, int n_in,
                              void* d_out, int out_size, void* d_ws, size_t ws_size,
                              hipStream_t stream) {
  const float* x = (const float*)d_in[0];
  const float *Wm[3], *bm[3], *U[3], *V[3], *bb[3];
  for (int l = 0; l < 3; ++l) {
    Wm[l] = (const float*)d_in[1 + 5 * l];
    bm[l] = (const float*)d_in[2 + 5 * l];
    U[l]  = (const float*)d_in[3 + 5 * l];
    V[l]  = (const float*)d_in[4 + 5 * l];
    bb[l] = (const float*)d_in[5 + 5 * l];
  }

  // Workspace carve (bf16 copies; ~57 MB total)
  char* ws = (char*)d_ws;
  size_t off = 0;
  auto carve = [&](size_t bytes) -> char* {
    char* p = ws + off;
    off += (bytes + 255) & ~(size_t)255;
    return p;
  };
  u16* actA = (u16*)carve((size_t)BATCH * DIN * 2);
  u16* actB = (u16*)carve((size_t)BATCH * DIN * 2);
  u16 *wmb[3], *ub[3], *vb[3];
  for (int l = 0; l < 3; ++l) {
    wmb[l] = (u16*)carve((size_t)4096 * DIN * 2);
    ub[l]  = (u16*)carve((size_t)RANK * DIN * 2);
    vb[l]  = (u16*)carve((size_t)DOUT * RANK * 2);
  }

  auto cvt = [&](const float* src, u16* dst, size_t n) {
    int n8 = (int)(n / 8);
    k_cvt_bf16<<<dim3((n8 + 255) / 256), dim3(256), 0, stream>>>(src, dst, n8);
  };
  cvt(x, actA, (size_t)BATCH * DIN);
  for (int l = 0; l < 3; ++l) {
    cvt(Wm[l], wmb[l], (size_t)4096 * DIN);
    cvt(U[l],  ub[l],  (size_t)RANK * DIN);
    cvt(V[l],  vb[l],  (size_t)DOUT * RANK);
  }

  dim3 grid(BATCH / TILE_M), block(128);
  k_apg_layer<<<grid, block, 0, stream>>>(actA, wmb[0], bm[0], ub[0], vb[0], bb[0],
                                          actB, nullptr, 1);
  k_apg_layer<<<grid, block, 0, stream>>>(actB, wmb[1], bm[1], ub[1], vb[1], bb[1],
                                          actA, nullptr, 1);
  k_apg_layer<<<grid, block, 0, stream>>>(actA, wmb[2], bm[2], ub[2], vb[2], bb[2],
                                          nullptr, (float*)d_out, 0);
  (void)in_sizes; (void)n_in; (void)out_size; (void)ws_size;
}
